// intent_predictor_66511863546288
// MI455X (gfx1250) — matrix operand order repaired and checked by reading, not compiled, not run on hardware
//
#include <hip/hip_runtime.h>

// ---------------------------------------------------------------------------
// Problem constants (fixed by the reference)
// ---------------------------------------------------------------------------
#define NN   4096      // nodes
#define FD   4096      // feature dim
#define EE   131072    // edges
#define OUTD 3

typedef __attribute__((ext_vector_type(16))) __bf16 v16bf;
typedef __attribute__((ext_vector_type(8)))  float  v8f;
typedef __attribute__((ext_vector_type(4)))  int    v4i_;

#define AS1 __attribute__((address_space(1)))
#define AS3 __attribute__((address_space(3)))

// CDNA5 async global->LDS copy path (guarded; falls back to register staging)
#if defined(__has_builtin)
#  if __has_builtin(__builtin_amdgcn_global_load_async_to_lds_b128)
#    define HAVE_ASYNC_LDS 1
#  endif
#endif
#ifndef HAVE_ASYNC_LDS
#  define HAVE_ASYNC_LDS 0
#endif

__device__ __forceinline__ void wait_async0() {
#if defined(__has_builtin) && __has_builtin(__builtin_amdgcn_s_wait_asynccnt)
    __builtin_amdgcn_s_wait_asynccnt(0);
#else
    asm volatile("s_wait_asynccnt 0x0" ::: "memory");
#endif
}

// fp32 -> bf16 round-to-nearest-even
__device__ __forceinline__ unsigned short f2bf(float f) {
    unsigned int u = __float_as_uint(f);
    u += 0x7FFFu + ((u >> 16) & 1u);
    return (unsigned short)(u >> 16);
}

struct __align__(8) US4 { unsigned short x, y, z, w; };

// ---------------------------------------------------------------------------
// fp32 -> bf16 conversion (row-major kept), 4 elements / thread
// ---------------------------------------------------------------------------
__global__ void cvt4_kernel(const float* __restrict__ in, unsigned short* __restrict__ out) {
    size_t i = ((size_t)blockIdx.x * blockDim.x + threadIdx.x) * 4;
    float4 v = *(const float4*)(in + i);
    US4 r; r.x = f2bf(v.x); r.y = f2bf(v.y); r.z = f2bf(v.z); r.w = f2bf(v.w);
    *(US4*)(out + i) = r;
}

// ---------------------------------------------------------------------------
// fused fp32 -> bf16 convert + transpose: out[c][r] = bf16(in[r][c])
// 32x32 LDS tile, coalesced reads and writes. Done once per weight matrix so
// the GEMM never needs scalar LDS scatter stores.
// ---------------------------------------------------------------------------
#define TT 32
__global__ void cvt_transpose_kernel(const float* __restrict__ in,
                                     unsigned short* __restrict__ out) {
    __shared__ unsigned short tile[TT][TT + 1];
    const int bx = blockIdx.x * TT;            // input col base
    const int by = blockIdx.y * TT;            // input row base
    const int tx = threadIdx.x & 31, ty = threadIdx.x >> 5;   // 32 x 8
    #pragma unroll
    for (int r = ty; r < TT; r += 8)
        tile[r][tx] = f2bf(in[(size_t)(by + r) * FD + bx + tx]);
    __syncthreads();
    #pragma unroll
    for (int r = ty; r < TT; r += 8)
        out[(size_t)(bx + r) * NN + by + tx] = tile[tx][r];
}

// ---------------------------------------------------------------------------
// Degree / dinv  (GCNConv improved=true: deg = indegree + 2, dinv = rsqrt)
// ---------------------------------------------------------------------------
__global__ void deg_count_kernel(const int* __restrict__ dst, int* __restrict__ deg) {
    int e = blockIdx.x * blockDim.x + threadIdx.x;
    atomicAdd(&deg[dst[e]], 1);
}

__global__ void dinv_kernel(const int* __restrict__ deg, float* __restrict__ dinv) {
    int n = blockIdx.x * blockDim.x + threadIdx.x;
    dinv[n] = rsqrtf((float)deg[n] + 2.0f);
}

// ---------------------------------------------------------------------------
// WMMA GEMM:  C[4096,4096] = A(bf16,row-major) @ B, with B given TRANSPOSED
// (BT[n][k]).  Block tile 128x128, BK=32, 4 waves, wave tile 64x64
// (4x4 wmma 16x16 f32 accumulators = 128 VGPRs).  Double-buffered LDS; both
// tiles are row-major [128][32] so staging is pure b128 (async-to-LDS when
// available) and all fragment reads are contiguous ds_load_b128.
// ---------------------------------------------------------------------------
#define BM  128
#define BN  128
#define BK  32
#define LDT 40    // padded LDS row stride (elements); 80B rows keep 16B align

__global__ __launch_bounds__(128, 1)
void gemm_bf16_wmma(const unsigned short* __restrict__ A,
                    const unsigned short* __restrict__ BT,
                    float* __restrict__ C,
                    const float* __restrict__ bias0,
                    const float* __restrict__ bias1) {
    __shared__ unsigned short As[2][BM * LDT];
    __shared__ unsigned short Bs[2][BN * LDT];

    const int K = FD, Nc = FD;
    const int tid  = threadIdx.x;        // 128 threads = 4 waves
    const int lane = tid & 31;
    const int wid  = tid >> 5;           // 0..3
    const int wm   = (wid >> 1) * 64;    // wave row offset in block tile
    const int wn   = (wid & 1) * 64;     // wave col offset in block tile
    const int bm   = blockIdx.y * BM;
    const int bn   = blockIdx.x * BN;

    // staging: each thread owns one full 32-wide k-slice of one row (64B)
    const unsigned short* gA = A  + (size_t)(bm + tid) * K;
    const unsigned short* gB = BT + (size_t)(bn + tid) * K;

    v8f acc[4][4];
    const v8f vzero = {0.f, 0.f, 0.f, 0.f, 0.f, 0.f, 0.f, 0.f};
    #pragma unroll
    for (int i = 0; i < 4; ++i)
        #pragma unroll
        for (int j = 0; j < 4; ++j) acc[i][j] = vzero;

    auto stage = [&](int buf, int k0) {
#if HAVE_ASYNC_LDS
        const unsigned short* sa = gA + k0;
        const unsigned short* sb = gB + k0;
        unsigned short* da = &As[buf][tid * LDT];
        unsigned short* db = &Bs[buf][tid * LDT];
        #pragma unroll
        for (int j = 0; j < 4; ++j) {
            __builtin_amdgcn_global_load_async_to_lds_b128(
                (AS1 v4i_*)(sa + j * 8), (AS3 v4i_*)(da + j * 8), 0, 0);
            __builtin_amdgcn_global_load_async_to_lds_b128(
                (AS1 v4i_*)(sb + j * 8), (AS3 v4i_*)(db + j * 8), 0, 0);
        }
#else
        const uint4* sa = (const uint4*)(gA + k0);
        const uint4* sb = (const uint4*)(gB + k0);
        uint4 ra0 = sa[0], ra1 = sa[1], ra2 = sa[2], ra3 = sa[3];
        uint4 rb0 = sb[0], rb1 = sb[1], rb2 = sb[2], rb3 = sb[3];
        uint4* da = (uint4*)&As[buf][tid * LDT];
        uint4* db = (uint4*)&Bs[buf][tid * LDT];
        da[0] = ra0; da[1] = ra1; da[2] = ra2; da[3] = ra3;
        db[0] = rb0; db[1] = rb1; db[2] = rb2; db[3] = rb3;
#endif
    };

    union Frag { v16bf v; uint4 q[2]; };

    auto compute = [&](int buf) {
        const int lm = lane & 15, lh = lane >> 4;
        Frag fb[4];
        #pragma unroll
        for (int nt = 0; nt < 4; ++nt) {
            // B fragment, contiguous k order used here (same as A): lanes lh=0 hold K 0..15, lh=1 hold K 16..31
            const uint4* p = (const uint4*)&Bs[buf][(wn + nt * 16 + lm) * LDT];
            fb[nt].q[0] = p[2 * lh];
            fb[nt].q[1] = p[2 * lh + 1];
        }
        #pragma unroll
        for (int mt = 0; mt < 4; ++mt) {
            // A fragment, contiguous k order used here (same as B): lanes lh=0 hold K 0..15, lh=1 hold K 16..31
            Frag fa;
            const uint4* p = (const uint4*)&As[buf][(wm + mt * 16 + lm) * LDT];
            fa.q[0] = p[2 * lh];
            fa.q[1] = p[2 * lh + 1];
            #pragma unroll
            for (int nt = 0; nt < 4; ++nt)
                acc[mt][nt] = __builtin_amdgcn_wmma_f32_16x16x32_bf16(
                    false, fa.v, false, fb[nt].v,
                    (short)0, acc[mt][nt], false, false);
        }
    };

    stage(0, 0);
#if HAVE_ASYNC_LDS
    wait_async0();
#endif
    __syncthreads();

    const int NT = K / BK;   // 128
    for (int t = 0; t < NT; ++t) {
        const int cur = t & 1;
        if (t + 1 < NT) stage(cur ^ 1, (t + 1) * BK);
        if (t + 2 < NT) {
            __builtin_prefetch(gA + (t + 2) * BK, 0, 1);
            __builtin_prefetch(gB + (t + 2) * BK, 0, 1);
        }
        compute(cur);
#if HAVE_ASYNC_LDS
        wait_async0();
#endif
        __syncthreads();
    }

    // Epilogue: C/D layout — VGPR r holds M = r + 8*lh, N = lane&15
    const int lm = lane & 15, lh = lane >> 4;
    #pragma unroll
    for (int nt = 0; nt < 4; ++nt) {
        const int col = bn + wn + nt * 16 + lm;
        float badd = 0.f;
        if (bias0) badd += bias0[col];
        if (bias1) badd += bias1[col];
        #pragma unroll
        for (int mt = 0; mt < 4; ++mt) {
            const int row0 = bm + wm + mt * 16 + lh * 8;
            #pragma unroll
            for (int r = 0; r < 8; ++r)
                C[(size_t)(row0 + r) * Nc + col] = acc[mt][nt][r] + badd;
        }
    }
}

// ---------------------------------------------------------------------------
// Sparse aggregation: agg[dst] += dinv[src]*dinv[dst] * xw[src]  (per edge)
// agg (64MB) and xw (64MB) both fit in the 192MB L2 -> atomics stay on-chip.
// ---------------------------------------------------------------------------
__global__ void aggregate_kernel(const int* __restrict__ src, const int* __restrict__ dst,
                                 const float* __restrict__ dinv,
                                 const float* __restrict__ xw, float* __restrict__ agg) {
    const int e = blockIdx.x;
    const int s = src[e], d = dst[e];
    const float coef = dinv[s] * dinv[d];
    const float* xs = xw + (size_t)s * FD;
    float* ad = agg + (size_t)d * FD;
    for (int i = threadIdx.x; i < FD; i += 256)
        atomicAdd(&ad[i], coef * xs[i]);
}

// out = relu(agg + 2*dinv^2 * xw + bias)
__global__ void gcn_post_kernel(const float* __restrict__ agg, const float* __restrict__ xw,
                                const float* __restrict__ dinv, const float* __restrict__ bias,
                                float* __restrict__ out) {
    size_t idx = (size_t)blockIdx.x * blockDim.x + threadIdx.x;
    const int n = (int)(idx >> 12);
    const int d = (int)(idx & (FD - 1));
    const float di = dinv[n];
    float v = agg[idx] + 2.f * di * di * xw[idx] + bias[d];
    out[idx] = v > 0.f ? v : 0.f;
}

// ---------------------------------------------------------------------------
// GCLSTM elementwise with H=C=0: Cn = sig(i)*tanh(c); H = sig(o + wc2*Cn)*tanh(Cn)
// ---------------------------------------------------------------------------
__device__ __forceinline__ float sigmoidf(float x) { return 1.f / (1.f + __expf(-x)); }

__global__ void lstm_elem_kernel(const float* __restrict__ pi, const float* __restrict__ pc,
                                 const float* __restrict__ po, const float* __restrict__ wc2,
                                 float* __restrict__ H) {
    size_t idx = (size_t)blockIdx.x * blockDim.x + threadIdx.x;
    const int d = (int)(idx & (FD - 1));
    const float Cn = sigmoidf(pi[idx]) * tanhf(pc[idx]);
    const float o  = sigmoidf(po[idx] + wc2[d] * Cn);
    H[idx] = o * tanhf(Cn);
}

// ---------------------------------------------------------------------------
// Head: out[row,:] = relu(H[row,:]) @ Wl + bl   (OUT=3, block-per-row reduce)
// ---------------------------------------------------------------------------
__global__ void final_linear_kernel(const float* __restrict__ H, const float* __restrict__ Wl,
                                    const float* __restrict__ bl, float* __restrict__ out) {
    const int row = blockIdx.x, tid = threadIdx.x;
    const float* h = H + (size_t)row * FD;
    float a0 = 0.f, a1 = 0.f, a2 = 0.f;
    for (int d = tid; d < FD; d += 256) {
        float v = h[d]; v = v > 0.f ? v : 0.f;
        a0 += v * Wl[d * 3 + 0];
        a1 += v * Wl[d * 3 + 1];
        a2 += v * Wl[d * 3 + 2];
    }
    __shared__ float red[3][256];
    red[0][tid] = a0; red[1][tid] = a1; red[2][tid] = a2;
    __syncthreads();
    for (int s = 128; s > 0; s >>= 1) {
        if (tid < s) {
            red[0][tid] += red[0][tid + s];
            red[1][tid] += red[1][tid + s];
            red[2][tid] += red[2][tid + s];
        }
        __syncthreads();
    }
    if (tid == 0) {
        out[row * 3 + 0] = red[0][0] + bl[0];
        out[row * 3 + 1] = red[1][0] + bl[1];
        out[row * 3 + 2] = red[2][0] + bl[2];
    }
}

// ---------------------------------------------------------------------------
// Launcher
// ---------------------------------------------------------------------------
extern "C" void kernel_launch(void* const* d_in, const int* in_sizes, int n_in,
                              void* d_out, int out_size, void* d_ws, size_t ws_size,
                              hipStream_t stream) {
    (void)in_sizes; (void)n_in; (void)out_size; (void)ws_size;

    const float* x    = (const float*)d_in[0];
    const int*   ei   = (const int*)d_in[1];
    const float* W1   = (const float*)d_in[2];
    const float* b1   = (const float*)d_in[3];
    const float* W2   = (const float*)d_in[4];
    const float* b2   = (const float*)d_in[5];
    const float* l1W  = (const float*)d_in[6];
    const float* l1b  = (const float*)d_in[8];
    const float* l1cb = (const float*)d_in[9];
    const float* l1wc = (const float*)d_in[10];
    const float* l2W  = (const float*)d_in[11];
    const float* l2b  = (const float*)d_in[13];
    const float* l2cb = (const float*)d_in[14];
    const float* l2wc = (const float*)d_in[15];
    const float* Wl   = (const float*)d_in[16];
    const float* bl   = (const float*)d_in[17];
    float* out = (float*)d_out;

    const int* esrc = ei;        // edge_index[0]
    const int* edst = ei + EE;   // edge_index[1]

    // workspace layout
    const size_t MAT = (size_t)NN * FD;            // elements
    char* w = (char*)d_ws;
    float* f0 = (float*)w;                 w += MAT * sizeof(float);  // activations / H
    float* f1 = (float*)w;                 w += MAT * sizeof(float);  // xw / pre_i
    float* f2 = (float*)w;                 w += MAT * sizeof(float);  // agg / pre_c
    float* f3 = (float*)w;                 w += MAT * sizeof(float);  // pre_o
    unsigned short* a_bf = (unsigned short*)w; w += MAT * sizeof(unsigned short);
    unsigned short* b_bf = (unsigned short*)w; w += MAT * sizeof(unsigned short); // W^T bf16
    int*   degi = (int*)w;                 w += NN * sizeof(int);
    float* dinv = (float*)w;

    const dim3 ggrid(FD / BN, NN / BM);            // 32 x 32 GEMM blocks
    const dim3 tgrid(FD / TT, NN / TT);            // 128 x 128 transpose blocks
    const int  cvtBlocks = (int)(MAT / 4 / 256);   // 16384
    const int  ewBlocks  = (int)(MAT / 256);       // 65536

    auto cvt = [&](const float* s, unsigned short* d) {
        cvt4_kernel<<<cvtBlocks, 256, 0, stream>>>(s, d);
    };
    auto cvtT = [&](const float* s, unsigned short* d) {
        cvt_transpose_kernel<<<tgrid, 256, 0, stream>>>(s, d);
    };
    auto gemm = [&](const unsigned short* A, const unsigned short* BT, float* C,
                    const float* bi0, const float* bi1) {
        gemm_bf16_wmma<<<ggrid, 128, 0, stream>>>(A, BT, C, bi0, bi1);
    };

    // ---- degrees ----
    (void)hipMemsetAsync(degi, 0, NN * sizeof(int), stream);
    deg_count_kernel<<<EE / 256, 256, 0, stream>>>(edst, degi);
    dinv_kernel<<<NN / 256, 256, 0, stream>>>(degi, dinv);

    // ---- GCN layer 1: f0 = relu(norm_agg(x @ W1) + b1) ----
    cvt(x, a_bf);
    cvtT(W1, b_bf);
    gemm(a_bf, b_bf, f1, nullptr, nullptr);                     // xw
    (void)hipMemsetAsync(f2, 0, MAT * sizeof(float), stream);
    aggregate_kernel<<<EE, 256, 0, stream>>>(esrc, edst, dinv, f1, f2);
    gcn_post_kernel<<<ewBlocks, 256, 0, stream>>>(f2, f1, dinv, b1, f0);

    // ---- GCN layer 2 ----
    cvt(f0, a_bf);
    cvtT(W2, b_bf);
    gemm(a_bf, b_bf, f1, nullptr, nullptr);
    (void)hipMemsetAsync(f2, 0, MAT * sizeof(float), stream);
    aggregate_kernel<<<EE, 256, 0, stream>>>(esrc, edst, dinv, f1, f2);
    gcn_post_kernel<<<ewBlocks, 256, 0, stream>>>(f2, f1, dinv, b2, f0);

    // ---- GCLSTM layer 1 (H=C=0: only i,c,o gates matter; cW GEMMs vanish) ----
    cvt(f0, a_bf);
    cvtT(l1W + 0 * MAT, b_bf);  gemm(a_bf, b_bf, f1, l1b + 0 * FD, l1cb + 0 * FD); // i
    cvtT(l1W + 2 * MAT, b_bf);  gemm(a_bf, b_bf, f2, l1b + 2 * FD, l1cb + 2 * FD); // c
    cvtT(l1W + 3 * MAT, b_bf);  gemm(a_bf, b_bf, f3, l1b + 3 * FD, l1cb + 3 * FD); // o
    lstm_elem_kernel<<<ewBlocks, 256, 0, stream>>>(f1, f2, f3, l1wc + 2 * FD, f0);

    // ---- GCLSTM layer 2 ----
    cvt(f0, a_bf);
    cvtT(l2W + 0 * MAT, b_bf);  gemm(a_bf, b_bf, f1, l2b + 0 * FD, l2cb + 0 * FD);
    cvtT(l2W + 2 * MAT, b_bf);  gemm(a_bf, b_bf, f2, l2b + 2 * FD, l2cb + 2 * FD);
    cvtT(l2W + 3 * MAT, b_bf);  gemm(a_bf, b_bf, f3, l2b + 3 * FD, l2cb + 3 * FD);
    lstm_elem_kernel<<<ewBlocks, 256, 0, stream>>>(f1, f2, f3, l2wc + 2 * FD, f0);

    // ---- head ----
    final_linear_kernel<<<NN, 256, 0, stream>>>(f0, Wl, bl, out);
}
